// ModuleM_17343077941703
// MI455X (gfx1250) — matrix operand order repaired and checked by reading, not compiled, not run on hardware
//
#include <hip/hip_runtime.h>

// ---------------------------------------------------------------------------
// Problem constants (fixed by the reference's setup_inputs)
// ---------------------------------------------------------------------------
constexpr int BATCH = 8;
constexpr int HGT   = 128;
constexpr int WID   = 256;
constexpr int CC    = 128;                 // left/right channels
constexpr int VOLC  = 160;                 // 134 input channels padded to 5*32
constexpr int NPIX  = BATCH * HGT * WID;   // 262144
constexpr float LEAK = 0.2f;

// ---------------------------------------------------------------------------
// Vector types (signatures match the gfx1250 WMMA builtins)
// ---------------------------------------------------------------------------
typedef __attribute__((ext_vector_type(16))) __bf16       v16bf;
typedef __attribute__((ext_vector_type(8)))  float        v8f;
typedef __attribute__((ext_vector_type(4)))  unsigned int v4u;
typedef __attribute__((ext_vector_type(2)))  unsigned int v2u;
typedef __attribute__((ext_vector_type(4)))  float        v4f;

union FragBF { v16bf v; v4u u4[2]; __bf16 h[16]; };
union Pack4  { __bf16 h[4]; v2u u; };

// ---------------------------------------------------------------------------
// Prep: build the 160-channel bf16 "vol" tensor
//   ch 0..127  : left (bf16)
//   ch 128..132: cost_i = mean_c(left) * warped(w + i - 2)   (SR = 2)
//   ch 133     : up (bilinear 2x upsample of prev_disp)
//   ch 134..159: zero pad
// One wave32 per pixel.
// ---------------------------------------------------------------------------
__device__ __forceinline__ float upsample_prev(const float* __restrict__ prev,
                                               int b, int h, int w)
{
    const int Hs = HGT / 2, Ws = WID / 2;
    float sy = (h + 0.5f) * 0.5f - 0.5f;     // half-pixel 2x upsample coords
    float sx = (w + 0.5f) * 0.5f - 0.5f;
    float y0f = floorf(sy), x0f = floorf(sx);
    float fy = sy - y0f, fx = sx - x0f;
    int y0 = (int)y0f, x0 = (int)x0f;
    int y1 = y0 + 1,   x1 = x0 + 1;
    y0 = min(max(y0, 0), Hs - 1); y1 = min(max(y1, 0), Hs - 1);
    x0 = min(max(x0, 0), Ws - 1); x1 = min(max(x1, 0), Ws - 1);
    const float* p = prev + (size_t)b * Hs * Ws;
    float v00 = p[y0 * Ws + x0], v01 = p[y0 * Ws + x1];
    float v10 = p[y1 * Ws + x0], v11 = p[y1 * Ws + x1];
    return (1.f - fy) * ((1.f - fx) * v00 + fx * v01)
         +        fy  * ((1.f - fx) * v10 + fx * v11);
}

// Faithful to the reference's pixel-flat gather: indices b*H*W + y*W + x read
// from right.reshape(-1) (flattened over all dims including channels).
__device__ __forceinline__ float warp_sample(const float* __restrict__ rightFlat,
                                             int b, float cx, float cy)
{
    float x0 = floorf(cx), x1 = x0 + 1.f;
    float y0 = floorf(cy), y1 = y0 + 1.f;
    float wx0 = x1 - cx, wx1 = cx - x0;
    float wy0 = y1 - cy, wy1 = cy - y0;
    const float xm = (float)(WID - 1), ym = (float)(HGT - 1);
    float x0s = fminf(fmaxf(x0, 0.f), xm), x1s = fminf(fmaxf(x1, 0.f), xm);
    float y0s = fminf(fmaxf(y0, 0.f), ym), y1s = fminf(fmaxf(y1, 0.f), ym);
    float base = (float)(b * HGT * WID);
    int i00 = (int)(x0s + base + y0s * (float)WID);
    int i01 = (int)(x0s + base + y1s * (float)WID);
    int i10 = (int)(x1s + base + y0s * (float)WID);
    int i11 = (int)(x1s + base + y1s * (float)WID);
    return wx0 * wy0 * rightFlat[i00] + wx0 * wy1 * rightFlat[i01]
         + wx1 * wy0 * rightFlat[i10] + wx1 * wy1 * rightFlat[i11];
}

__global__ __launch_bounds__(256) void prep_kernel(
    const float* __restrict__ left, const float* __restrict__ right,
    const float* __restrict__ prev, __bf16* __restrict__ vol)
{
    const int lane = threadIdx.x & 31;
    const int wv   = threadIdx.x >> 5;
    const int pix  = blockIdx.x * 8 + wv;          // 8 waves -> 8 pixels/block
    const int w = pix % WID;
    const int h = (pix / WID) % HGT;
    const int b =  pix / (WID * HGT);

    // convert 4 left channels per lane, accumulate for the channel mean
    v4f lv = *(const v4f*)(left + (size_t)pix * CC + lane * 4);
    Pack4 pk;
    pk.h[0] = (__bf16)lv[0]; pk.h[1] = (__bf16)lv[1];
    pk.h[2] = (__bf16)lv[2]; pk.h[3] = (__bf16)lv[3];
    __bf16* vp = vol + (size_t)pix * VOLC;
    *(v2u*)(vp + lane * 4) = pk.u;

    float s = lv[0] + lv[1] + lv[2] + lv[3];
    for (int off = 16; off > 0; off >>= 1) s += __shfl_xor(s, off, 32);
    const float leftmean = s * (1.0f / (float)CC);

    if (lane < 5) {                                 // cost channels 128..132
        const int wq = w + lane - 2;
        float c = 0.0f;
        if (wq >= 0 && wq < WID) {
            float u = upsample_prev(prev, b, h, wq);
            c = warp_sample(right, b, (float)wq - u, (float)h) * leftmean;
        }
        vp[CC + lane] = (__bf16)c;
    } else if (lane == 5) {                         // channel 133 = up
        vp[CC + 5] = (__bf16)upsample_prev(prev, b, h, w);
    } else {                                        // channels 134..159 = 0
        vp[CC + lane] = (__bf16)0.0f;
    }
}

// ---------------------------------------------------------------------------
// Weight repack: HWIO fp32 -> per-(tap, kChunk, nChunk) bf16 WMMA B-fragments.
// B fragment k order used here (same as A): lane&15 = N column; halves 0..15 hold
// K = (lane>>4)*16 + h. Stored lane-major, 32 bytes/lane (1 KiB per fragment)
// so the conv inner loop loads each fragment with two coalesced b128 reads.
// ---------------------------------------------------------------------------
__global__ void repack_weights(const float* __restrict__ w, __bf16* __restrict__ frag,
                               int Cin, int CinP, int Cout, int CoutP)
{
    const int KC = CinP / 32, NC = CoutP / 16;
    const int total = 9 * KC * NC * 32;
    const int t = blockIdx.x * blockDim.x + threadIdx.x;
    if (t >= total) return;
    const int lane = t & 31;
    const int f    = t >> 5;
    const int nc   = f % NC;
    const int r    = f / NC;
    const int kc   = r % KC;
    const int tap  = r / KC;
    const int n     = lane & 15;
    const int co    = nc * 16 + n;
    const int kbase = kc * 32 + (lane >> 4) * 16;
    FragBF fr;
#pragma unroll
    for (int hi = 0; hi < 16; ++hi) {
        const int ci = kbase + hi;
        float v = 0.0f;
        if (ci < Cin && co < Cout) v = w[((size_t)tap * Cin + ci) * Cout + co];
        fr.h[hi] = (__bf16)v;
    }
    v4u* dst = (v4u*)(frag + (size_t)f * 512 + lane * 16);
    dst[0] = fr.u4[0];
    dst[1] = fr.u4[1];
}

// ---------------------------------------------------------------------------
// Fused 3x3 SAME conv (+bias, +leaky-relu) as implicit GEMM on WMMA bf16.
// Block tile: 16 consecutive pixels (one row) x COUT channels.
// One wave per 16 out-channels; input halo tile (3 rows x 18 px x CIN bf16)
// staged in LDS and shared by all waves. Row stride padded by 8 halves so
// b128 A-fragment reads are LDS-bank-conflict free.
// ---------------------------------------------------------------------------
template<int CIN, int COUT, bool RELU, bool FINAL>
__global__ __launch_bounds__(32 * (COUT / 16)) void conv3x3_wmma(
    const __bf16* __restrict__ in, const __bf16* __restrict__ wfrag,
    const float* __restrict__ bias, int coutReal,
    __bf16* __restrict__ out, float* __restrict__ fout)
{
    constexpr int KC = CIN / 32;
    constexpr int NC = COUT / 16;
    constexpr int LS = CIN + 8;              // padded LDS row stride (halves)
    __shared__ __bf16 lds[3 * 18 * LS];

    const int tid  = threadIdx.x;
    const int lane = tid & 31;
    const int wave = tid >> 5;

    const int w0 = blockIdx.x * 16;
    const int h  = blockIdx.y;
    const int b  = blockIdx.z;

    // cooperative halo-tile load: 3 rows x 18 px x CIN halves (zeros at edges)
    constexpr int SEG = CIN / 8;             // 16-byte segments per pixel
    for (int s = tid; s < 3 * 18 * SEG; s += 32 * NC) {
        const int ci  = (s % SEG) * 8;
        const int p   = s / SEG;
        const int px  = p % 18;
        const int row = p / 18;
        const int hh  = h + row - 1;
        const int ww  = w0 - 1 + px;
        v4u val = {0u, 0u, 0u, 0u};
        if (hh >= 0 && hh < HGT && ww >= 0 && ww < WID)
            val = *(const v4u*)(in + (((size_t)b * HGT + hh) * WID + ww) * CIN + ci);
        *(v4u*)(&lds[(row * 18 + px) * LS + ci]) = val;
    }
    __syncthreads();

    // A k order used here (same as B): lane&15 = M (pixel); halves 0..15 -> K=kb..kb+15
    // with kb = (lane>>4)*16.
    const int m  = lane & 15;
    const int kb = (lane >> 4) * 16;

    v8f acc = {};
    for (int tap = 0; tap < 9; ++tap) {
        const int ky = tap / 3, kx = tap % 3;
        const __bf16* arow = &lds[(ky * 18 + m + kx) * LS + kb];
#pragma unroll
        for (int kc = 0; kc < KC; ++kc) {
            FragBF a, bq;
            const __bf16* ap = arow + kc * 32;
            a.u4[0] = *(const v4u*)ap;            // K = kb .. kb+7
            a.u4[1] = *(const v4u*)(ap + 8);      // K = kb+8 .. kb+15
            const __bf16* bp = wfrag + ((size_t)(tap * KC + kc) * NC + wave) * 512
                                     + lane * 16;
            bq.u4[0] = *(const v4u*)bp;
            bq.u4[1] = *(const v4u*)(bp + 8);
            acc = __builtin_amdgcn_wmma_f32_16x16x32_bf16(
                      false, a.v, false, bq.v, (short)0, acc, false, false);
        }
    }

    // C/D 16x16 f32 layout: N = lane&15; M = (lane>>4)*8 + vgpr
    const int n  = lane & 15;
    const int co = wave * 16 + n;
    const float bv = (co < coutReal) ? bias[co] : 0.0f;
#pragma unroll
    for (int v = 0; v < 8; ++v) {
        const int mm = (lane >> 4) * 8 + v;       // pixel within tile
        float x = acc[v] + bv;
        if (RELU) x = (x >= 0.0f) ? x : LEAK * x;
        const int w = w0 + mm;
        const size_t pix = ((size_t)b * HGT + h) * WID + w;
        if (!FINAL)           out[pix * COUT + co] = (__bf16)x;
        else if (co == 0)     fout[pix] = x;      // 1-channel linear head
    }
}

// ---------------------------------------------------------------------------
// Launch
// ---------------------------------------------------------------------------
extern "C" void kernel_launch(void* const* d_in, const int* in_sizes, int n_in,
                              void* d_out, int out_size, void* d_ws, size_t ws_size,
                              hipStream_t stream)
{
    const float* left  = (const float*)d_in[0];
    const float* right = (const float*)d_in[1];
    const float* prev  = (const float*)d_in[2];
    const float* k[6];  const float* bias[6];
    for (int i = 0; i < 6; ++i) {
        k[i]    = (const float*)d_in[3 + 2 * i];
        bias[i] = (const float*)d_in[4 + 2 * i];
    }

    char* ws = (char*)d_ws;
    size_t off = 0;
    auto alloc = [&](size_t bytes) -> char* {
        char* p = ws + off;
        off += (bytes + 1023) & ~(size_t)1023;
        return p;
    };
    __bf16* vol = (__bf16*)alloc((size_t)NPIX * VOLC * 2);   //  84.0 MB
    __bf16* a1  = (__bf16*)alloc((size_t)NPIX * 128 * 2);    //  67.1 MB
    __bf16* a2  = (__bf16*)alloc((size_t)NPIX * 128 * 2);    //  67.1 MB
    __bf16* a3  = (__bf16*)alloc((size_t)NPIX *  96 * 2);    //  50.3 MB
    __bf16* a4  = (__bf16*)alloc((size_t)NPIX *  64 * 2);    //  33.6 MB
    __bf16* a5  = (__bf16*)alloc((size_t)NPIX *  32 * 2);    //  16.8 MB

    const int CinP[6]  = {160, 128, 128, 96, 64, 32};
    const int Cin[6]   = {134, 128, 128, 96, 64, 32};
    const int Cout[6]  = {128, 128,  96, 64, 32,  1};
    const int CoutP[6] = {128, 128,  96, 64, 32, 16};
    __bf16* wf[6];
    for (int i = 0; i < 6; ++i) {
        const size_t frags = (size_t)9 * (CinP[i] / 32) * (CoutP[i] / 16);
        wf[i] = (__bf16*)alloc(frags * 1024);
        const int total = (int)frags * 32;
        repack_weights<<<(total + 255) / 256, 256, 0, stream>>>(
            k[i], wf[i], Cin[i], CinP[i], Cout[i], CoutP[i]);
    }

    prep_kernel<<<NPIX / 8, 256, 0, stream>>>(left, right, prev, vol);

    const dim3 grid(WID / 16, HGT, BATCH);
    conv3x3_wmma<160, 128, true,  false><<<grid, 256, 0, stream>>>(vol, wf[0], bias[0], 128, a1, nullptr);
    conv3x3_wmma<128, 128, true,  false><<<grid, 256, 0, stream>>>(a1,  wf[1], bias[1], 128, a2, nullptr);
    conv3x3_wmma<128,  96, true,  false><<<grid, 192, 0, stream>>>(a2,  wf[2], bias[2],  96, a3, nullptr);
    conv3x3_wmma< 96,  64, true,  false><<<grid, 128, 0, stream>>>(a3,  wf[3], bias[3],  64, a4, nullptr);
    conv3x3_wmma< 64,  32, true,  false><<<grid,  64, 0, stream>>>(a4,  wf[4], bias[4],  32, a5, nullptr);
    conv3x3_wmma< 32,  16, false, true ><<<grid,  32, 0, stream>>>(a5,  wf[5], bias[5],   1, nullptr, (float*)d_out);

    (void)in_sizes; (void)n_in; (void)out_size; (void)ws_size;
}
